// MPNN_72275709657227
// MI455X (gfx1250) — matrix operand order repaired and checked by reading, not compiled, not run on hardware
//
#include <hip/hip_runtime.h>
#include <hip/hip_bf16.h>
#include <math.h>

// MPNN: B=8, N=1024, H=32.
// Edge MLP (2->32->32->1) over 8.4M edges dominates (~19 GFLOP vs ~4MB traffic):
// compute-bound -> map the 32x32 hidden layer onto v_wmma_f32_16x16x32_f16
// (K=32 matches exactly; one WMMA = full K reduction for a 16x16 tile).
// Round 2: cross-lane reductions via DPP16 VALU (no ds_bpermute/dscnt stalls),
// branch-free tanh (v_exp_f32 + v_rcp_f32), no per-row A shuffle.

typedef __attribute__((ext_vector_type(16))) _Float16 v16h;
typedef __attribute__((ext_vector_type(8)))  float    v8f;

#define NNODES 1024
#define BATCH  8

__device__ __forceinline__ float leaky(float x) {
    // leaky_relu slope 0.01: max(x, 0.01x) is exact for both signs
    return fmaxf(x, 0.01f * x);
}

// Branch-free tanh: (e^{2x}-1)/(e^{2x}+1) with clamp so e never overflows.
// v_exp_f32 (exp2) + v_rcp_f32, ~8 straight-line VALU ops, no exec masking.
__device__ __forceinline__ float fast_tanh(float x) {
    float xc = fminf(fmaxf(x, -15.f), 15.f);
    float e2 = __builtin_amdgcn_exp2f(xc * 2.885390081777927f); // exp(2x)
    return (e2 - 1.f) * __builtin_amdgcn_rcpf(e2 + 1.f);
}

// x + (x from lane selected by DPP16 ctrl). Stays within 16-lane rows.
template <int CTRL>
__device__ __forceinline__ float dpp_add(float x) {
    int y = __builtin_amdgcn_mov_dpp(__float_as_int(x), CTRL, 0xf, 0xf, true);
    return x + __int_as_float(y);
}

// Sum over each 16-lane half (halves independent): xor1, xor2, then
// half-mirror (==xor4 once quads are uniform) and mirror (==xor8).
__device__ __forceinline__ float reduce16(float t) {
    t = dpp_add<0xB1>(t);   // quad_perm [1,0,3,2]  : xor 1
    t = dpp_add<0x4E>(t);   // quad_perm [2,3,0,1]  : xor 2
    t = dpp_add<0x141>(t);  // row_half_mirror      : xor 4 (quads uniform)
    t = dpp_add<0x140>(t);  // row_mirror           : xor 8 (octs uniform)
    return t;
}

// Per-lane constant fragments for a 2->32->32->1 MLP evaluated 16 rows/tile.
// Layer-1 weights are held per-lane in the K-slot order of the 16-bit 16x32
// A-fragment so activations are produced directly in WMMA layout:
//   lane<16 : K = {0..7,16..23};  lane>=16 : K = {8..15,24..31}
// Layer-2 W (32x32) is prepacked into two 32x16 B fragments (cols 0-15, 16-31):
//   same K-slot order as the A-fragment above, column n = lane&15.
struct MlpFrag {
    float w10[16], w11[16], b1v[16];
    v16h  b0f, b1f;
    float b2a, b2b, w3a, w3b, b3s;
};

__device__ __forceinline__ void mlp_setup(MlpFrag& f, int lane,
                                          const float* __restrict__ W1,
                                          const float* __restrict__ b1,
                                          const float* __restrict__ W2,
                                          const float* __restrict__ b2,
                                          const float* __restrict__ W3,
                                          const float* __restrict__ b3) {
    const int hi = (lane >= 16) ? 1 : 0;
    const int n  = lane & 15;
#pragma unroll
    for (int e = 0; e < 16; ++e) {
        const int k  = ((e < 8) ? e : e + 8) + hi * 8;   // A-fragment K slot
        f.w10[e] = W1[k];          // W1[0][k]
        f.w11[e] = W1[32 + k];     // W1[1][k]
        f.b1v[e] = b1[k];
        const int kb = k;                                // B-fragment K slot
        f.b0f[e] = (_Float16)W2[kb * 32 + n];            // cols 0..15
        f.b1f[e] = (_Float16)W2[kb * 32 + 16 + n];       // cols 16..31
    }
    f.b2a = b2[n];      f.b2b = b2[16 + n];
    f.w3a = W3[n];      f.w3b = W3[16 + n];
    f.b3s = b3[0];
}

// Evaluate layers 1+2 for a tile of 16 rows. Each lane owns one row
// (m = lane&15): e0/e1 are that row's two input features. Produces raw f32
// accumulators for output cols 0-15 (c0) and 16-31 (c1) in C/D layout
// (VGPR r -> row r (lanes 0-15) / r+8 (lanes 16-31), col = lane&15).
__device__ __forceinline__ void mlp_tile(const MlpFrag& f, float e0, float e1,
                                         v8f& c0, v8f& c1) {
    v16h a;
#pragma unroll
    for (int e = 0; e < 16; ++e) {
        float h = fmaf(e0, f.w10[e], fmaf(e1, f.w11[e], f.b1v[e]));
        a[e] = (_Float16)leaky(h);
    }
    v8f z = {};
    // (neg_a, A, neg_b, B, c_mod, C, reuse_a, reuse_b)
    c0 = __builtin_amdgcn_wmma_f32_16x16x32_f16(false, a, false, f.b0f,
                                                (short)0, z, false, false);
    c1 = __builtin_amdgcn_wmma_f32_16x16x32_f16(false, a, false, f.b1f,
                                                (short)0, z, false, false);
}

// Kernel 1: one wave per (b,i). Loop j in tiles of 16 edges, run edge MLP via
// WMMA, finish layer 3 with a DPP reduction per output row, weight by A[i,j],
// accumulate msg_sum[b,i].
__global__ void __launch_bounds__(256)
mpnn_edge_kernel(const float* __restrict__ x,   const float* __restrict__ A,
                 const float* __restrict__ Wm1, const float* __restrict__ bm1,
                 const float* __restrict__ Wm2, const float* __restrict__ bm2,
                 const float* __restrict__ Wm3, const float* __restrict__ bm3,
                 float* __restrict__ msg_sum) {
    const int lane = threadIdx.x & 31;
    const int col  = lane & 15;
    const int wid  = (blockIdx.x * blockDim.x + threadIdx.x) >> 5; // 0..8191
    const int b    = wid >> 10;
    const int i    = wid & (NNODES - 1);

    MlpFrag f;
    mlp_setup(f, lane, Wm1, bm1, Wm2, bm2, Wm3, bm3);

    const float2* xv = (const float2*)x;          // x[b][n] = {rho, V}
    const float2  xi = xv[b * NNODES + i];
    const float rho_i = xi.x, V_i = xi.y;
    const float* Arow = A + (size_t)i * NNODES;

    float acc = 0.f;
#pragma unroll 2
    for (int j0 = 0; j0 < NNODES; j0 += 16) {
        const float2 xj = xv[b * NNODES + j0 + col];  // edge owned by this lane
        const float e0 = rho_i + xj.x;
        const float e1 = V_i   + xj.y;
        const float Aval = Arow[j0 + col];
        if (j0 + 64 < NNODES)                         // warm L2/WGP$ for A row
            __builtin_prefetch(&Arow[j0 + 64 + col], 0, 1);

        v8f c0, c1;
        mlp_tile(f, e0, e1, c0, c1);

#pragma unroll
        for (int r = 0; r < 8; ++r) {
            // bias + leaky on layer-2 outputs, then layer-3 partial dot
            float v0 = leaky(c0[r] + f.b2a);
            float v1 = leaky(c1[r] + f.b2b);
            float t  = fmaf(v0, f.w3a, v1 * f.w3b);
            // reduce the 16 columns held in each half (pure DPP VALU):
            // lanes 0-15 -> dot for row r, lanes 16-31 -> row r+8
            t = reduce16(t);
            const float me = fast_tanh(t + f.b3s);
            // the lane already holding column r owns A for that row: no shuffle
            if (col == r) acc = fmaf(me, Aval, acc);
        }
    }
    // acc is spread over all 32 lanes (8 row-slots x 2 halves): final reduce
    acc = reduce16(acc);
    acc += __shfl_xor(acc, 16, 32);                   // cross-half, once
    if (lane == 0) msg_sum[wid] = acc;
}

// Kernel 2: node MLP on (rho, msg_sum). Only 8192 rows; one wave per 16 nodes,
// same WMMA tile path, write tanh output.
__global__ void __launch_bounds__(256)
mpnn_node_kernel(const float* __restrict__ x,   const float* __restrict__ msg_sum,
                 const float* __restrict__ Wx1, const float* __restrict__ bx1,
                 const float* __restrict__ Wx2, const float* __restrict__ bx2,
                 const float* __restrict__ Wx3, const float* __restrict__ bx3,
                 float* __restrict__ out) {
    const int lane = threadIdx.x & 31;
    const int col  = lane & 15;
    const int wid  = (blockIdx.x * blockDim.x + threadIdx.x) >> 5; // 0..511
    const int base = wid * 16;

    MlpFrag f;
    mlp_setup(f, lane, Wx1, bx1, Wx2, bx2, Wx3, bx3);

    const float2* xv = (const float2*)x;
    const int node = base + col;
    const float e0 = xv[node].x;        // rho
    const float e1 = msg_sum[node];

    v8f c0, c1;
    mlp_tile(f, e0, e1, c0, c1);

#pragma unroll
    for (int r = 0; r < 8; ++r) {
        float v0 = leaky(c0[r] + f.b2a);
        float v1 = leaky(c1[r] + f.b2b);
        float t  = fmaf(v0, f.w3a, v1 * f.w3b);
        t = reduce16(t);
        const float o = fast_tanh(t + f.b3s);
        const int m = r + ((lane >> 4) << 3);     // r (half0) or r+8 (half1)
        if (col == r) out[base + m] = o;          // lanes r and 16+r write
    }
}

extern "C" void kernel_launch(void* const* d_in, const int* in_sizes, int n_in,
                              void* d_out, int out_size, void* d_ws, size_t ws_size,
                              hipStream_t stream) {
    const float* x   = (const float*)d_in[0];
    const float* A   = (const float*)d_in[1];
    const float* Wm1 = (const float*)d_in[2];
    const float* bm1 = (const float*)d_in[3];
    const float* Wm2 = (const float*)d_in[4];
    const float* bm2 = (const float*)d_in[5];
    const float* Wm3 = (const float*)d_in[6];
    const float* bm3 = (const float*)d_in[7];
    const float* Wx1 = (const float*)d_in[8];
    const float* bx1 = (const float*)d_in[9];
    const float* Wx2 = (const float*)d_in[10];
    const float* bx2 = (const float*)d_in[11];
    const float* Wx3 = (const float*)d_in[12];
    const float* bx3 = (const float*)d_in[13];

    float* msg_sum = (float*)d_ws;                 // B*N floats = 32 KB
    float* out     = (float*)d_out;                // B*N floats (B,N,1)

    // 8192 waves (one per (b,i)), 8 waves / 256-thread block
    mpnn_edge_kernel<<<(BATCH * NNODES) / 8, 256, 0, stream>>>(
        x, A, Wm1, bm1, Wm2, bm2, Wm3, bm3, msg_sum);

    // 512 waves (16 nodes each)
    mpnn_node_kernel<<<(BATCH * NNODES / 16) / 8, 256, 0, stream>>>(
        x, msg_sum, Wx1, bx1, Wx2, bx2, Wx3, bx3, out);
}
